// CFRMClassifier_39831526703830
// MI455X (gfx1250) — matrix operand order repaired and checked by reading, not compiled, not run on hardware
//
#include <hip/hip_runtime.h>
#include <cmath>
#include <stdint.h>

// Problem constants (match reference)
#define H_DIM   512
#define B_DIM   64
#define T_DIM   1024
#define G3H     1536          // 3*H
#define C_DIM   32
#define NCLS    1000
#define NCLS_PAD 1024
#define KCLS    16448         // C*(H+2)
#define CW      514           // H+2
#define DECAY_F 0.85f
#define NWG_SCAN 8

typedef __bf16 bf16_t;
typedef __attribute__((ext_vector_type(16))) __bf16        v16bf;
typedef __attribute__((ext_vector_type(8)))  float         v8f;
typedef __attribute__((ext_vector_type(8)))  unsigned int  v8u;

// ---------------- bf16 helpers (RNE) ----------------
__device__ inline unsigned short bf_bits(float f) {
  unsigned u = __builtin_bit_cast(unsigned, f);
  unsigned r = u + 0x7FFFu + ((u >> 16) & 1u);
  return (unsigned short)(r >> 16);
}
__device__ inline bf16_t to_bf16(float f) {
  unsigned short b = bf_bits(f);
  return __builtin_bit_cast(bf16_t, b);
}
__device__ inline float sigmoidf_(float x) { return 1.f / (1.f + __expf(-x)); }

// ---------------- WMMA fragment loaders ----------------
// A: 16x32 bf16 tile, row-major [M,K] source (row stride ldk elems, 16B aligned).
// ISA layout: lanes0-15 (g=0): V0..V3 = K0..7, V4..V7 = K16..23
//             lanes16-31 (g=1): V0..V3 = K8..15, V4..V7 = K24..31
__device__ inline v16bf load_a_tile(const bf16_t* __restrict__ tile, int ldk, int lane) {
  const int m = lane & 15, g = lane >> 4;
  const uint4* p = (const uint4*)(tile + (size_t)m * ldk);
  uint4 d0 = p[g];         // dwords 4g..4g+3  (K pairs 8g..8g+7)
  uint4 d1 = p[2 + g];     // dwords 8+4g..    (K pairs 16+8g..)
  v8u d = {d0.x, d0.y, d0.z, d0.w, d1.x, d1.y, d1.z, d1.w};
  return __builtin_bit_cast(v16bf, d);
}
// B: 32x16 bf16 tile taken from a row-major [N,K] weight (i.e. B^T layout).
// Lane ln = output column n; g selects K 8g..8g+7 and 16+8g..16+8g+7 (same as A).
__device__ inline v16bf load_b_tile(const bf16_t* __restrict__ tile, int ldk, int lane) {
  const int n = lane & 15, g = lane >> 4;
  const uint4* p = (const uint4*)(tile + (size_t)n * ldk);
  uint4 d0 = p[g], d1 = p[2 + g];
  v8u d = {d0.x, d0.y, d0.z, d0.w, d1.x, d1.y, d1.z, d1.w};
  return __builtin_bit_cast(v16bf, d);
}
__device__ inline v8f wmma_bf16(v16bf a, v16bf b, v8f c) {
  return __builtin_amdgcn_wmma_f32_16x16x32_bf16(false, a, false, b, (short)0, c,
                                                 false, false);
}

// ---------------- device-wide barrier (monotonic counter) ----------------
__device__ inline void grid_barrier(unsigned* bar, unsigned step) {
  __threadfence();
  __syncthreads();
  if (threadIdx.x == 0) {
    atomicAdd(bar, 1u);
    const unsigned target = (step + 1u) * NWG_SCAN;
    while (__hip_atomic_load(bar, __ATOMIC_ACQUIRE, __HIP_MEMORY_SCOPE_AGENT) < target)
      __builtin_amdgcn_s_sleep(4);
  }
  __syncthreads();
  __threadfence();
}

// ---------------- kernels ----------------

// Embedding gather + f32->bf16, 4 elems/thread
__global__ __launch_bounds__(256) void gather_cvt_kernel(
    const int* __restrict__ tokens, const float* __restrict__ emb,
    bf16_t* __restrict__ xb) {
  const long long i = (long long)blockIdx.x * blockDim.x + threadIdx.x;
  const int bt = (int)(i >> 7);                 // H/4 = 128 quads per row
  if (bt >= B_DIM * T_DIM) return;
  const int h4 = ((int)i & 127) << 2;
  const int tok = tokens[bt];
  const float4 f = *(const float4*)(emb + (size_t)tok * H_DIM + h4);
  uint2 o;
  o.x = ((unsigned)bf_bits(f.y) << 16) | bf_bits(f.x);
  o.y = ((unsigned)bf_bits(f.w) << 16) | bf_bits(f.z);
  *(uint2*)(xb + (size_t)bt * H_DIM + h4) = o;
}

// Generic f32 -> bf16, 4 elems/thread (count divisible by 4)
__global__ __launch_bounds__(256) void cvt_bf16_kernel(
    const float* __restrict__ in, bf16_t* __restrict__ out, long long n4) {
  const long long i = (long long)blockIdx.x * blockDim.x + threadIdx.x;
  if (i >= n4) return;
  const float4 f = ((const float4*)in)[i];
  uint2 o;
  o.x = ((unsigned)bf_bits(f.y) << 16) | bf_bits(f.x);
  o.y = ((unsigned)bf_bits(f.w) << 16) | bf_bits(f.z);
  ((uint2*)out)[i] = o;
}

// f32 -> bf16 with zero row padding (for wcls -> [1024, KCLS])
__global__ __launch_bounds__(256) void cvt_pad_kernel(
    const float* __restrict__ in, bf16_t* __restrict__ out, int rows, int cols,
    int padRows) {
  const long long i = (long long)blockIdx.x * blockDim.x + threadIdx.x;
  const int q = cols >> 2;
  const long long total = (long long)padRows * q;
  if (i >= total) return;
  const int r = (int)(i / q);
  const int c4 = ((int)(i % q)) << 2;
  float4 f = make_float4(0.f, 0.f, 0.f, 0.f);
  if (r < rows) f = *(const float4*)(in + (size_t)r * cols + c4);
  uint2 o;
  o.x = ((unsigned)bf_bits(f.y) << 16) | bf_bits(f.x);
  o.y = ((unsigned)bf_bits(f.w) << 16) | bf_bits(f.z);
  *(uint2*)(out + (size_t)r * cols + c4) = o;
}

// Generic WMMA GEMM: out[M,N] = A[M,K](bf16) * Bm[N,K](bf16)^T (+bias)
// Block = 8 waves; wave (w&3) -> m-tile, (w>>2) -> 64-col n-slab (4 tiles).
// Block tile 64x128. Two-stage ping-pong pipeline with fixed register names:
// no fragment copies, distance-2 load prefetch ahead of the consuming WMMA.
// Requires ksteps even (all K used are multiples of 64).
// mode 0: f32 out (xp). mode 1: bf16 out into interleaved summary layout
// (centers). mode 2: f32 out with n<outN guard (classifier).
__global__ __launch_bounds__(256) void gemm_bf16_kernel(
    const bf16_t* __restrict__ A, const bf16_t* __restrict__ Bm,
    const float* __restrict__ bias, void* __restrict__ out, int M, int N, int K,
    int mode, float outScale, float biasScale, int outN) {
  const int lane = threadIdx.x & 31;
  const int w = threadIdx.x >> 5;
  const int g = lane >> 4, ln = lane & 15;
  const int m0 = blockIdx.x * 64 + (w & 3) * 16;
  const int n0 = blockIdx.y * 128 + (w >> 2) * 64;
  const int ksteps = K >> 5;     // even
  v8f acc[4] = {};
  const bf16_t* aRow = A + (size_t)m0 * K;

  v16bf a0 = load_a_tile(aRow, K, lane);
  v16bf a1 = load_a_tile(aRow + 32, K, lane);
  v16bf b0[4], b1[4];
#pragma unroll
  for (int nt = 0; nt < 4; ++nt) {
    const bf16_t* br = Bm + (size_t)(n0 + nt * 16) * K;
    b0[nt] = load_b_tile(br, K, lane);
    b1[nt] = load_b_tile(br + 32, K, lane);
  }

  for (int ks = 0; ks < ksteps; ks += 2) {
    // consume stage 0 (k-step ks), refill with k-step ks+2
#pragma unroll
    for (int nt = 0; nt < 4; ++nt) acc[nt] = wmma_bf16(a0, b0[nt], acc[nt]);
    if (ks + 2 < ksteps) {
      a0 = load_a_tile(aRow + (ks + 2) * 32, K, lane);
#pragma unroll
      for (int nt = 0; nt < 4; ++nt)
        b0[nt] = load_b_tile(Bm + (size_t)(n0 + nt * 16) * K + (ks + 2) * 32, K, lane);
    }
    // consume stage 1 (k-step ks+1), refill with k-step ks+3
#pragma unroll
    for (int nt = 0; nt < 4; ++nt) acc[nt] = wmma_bf16(a1, b1[nt], acc[nt]);
    if (ks + 3 < ksteps) {
      a1 = load_a_tile(aRow + (ks + 3) * 32, K, lane);
#pragma unroll
      for (int nt = 0; nt < 4; ++nt)
        b1[nt] = load_b_tile(Bm + (size_t)(n0 + nt * 16) * K + (ks + 3) * 32, K, lane);
    }
  }

#pragma unroll
  for (int nt = 0; nt < 4; ++nt) {
    const int n = n0 + nt * 16 + ln;
    const float bv = (bias && n < outN) ? bias[n] : 0.f;
#pragma unroll
    for (int v = 0; v < 8; ++v) {
      const int m = m0 + v + 8 * g;
      const float val = acc[nt][v] * outScale + bv * biasScale;
      if (mode == 0)
        ((float*)out)[(size_t)m * N + n] = val;
      else if (mode == 1)
        ((bf16_t*)out)[(size_t)m * KCLS + (n >> 9) * CW + (n & 511)] = to_bf16(val);
      else if (n < outN)
        ((float*)out)[(size_t)m * outN + n] = val;
    }
  }
}

// Persistent GRU scan. 8 WGs, each owns a 64-col h slice (j0..j0+63) and the
// matching r/z/n columns of hp. Device barrier per step. Fused accumulation of
// hw (decay-weighted), hsum, and (WG0) the sigmoid spread projection.
__global__ __launch_bounds__(256) void gru_scan_kernel(
    const float* __restrict__ xp,          // [B*T, 3H] f32 (b-major, then t)
    const bf16_t* __restrict__ whh,        // [3H, H] bf16
    const float* __restrict__ bhh,         // [3H]
    const bf16_t* __restrict__ wsm,        // [C, H] bf16
    const float* __restrict__ bsv,         // [C]
    float* __restrict__ hf,                // [2][B*H] f32 ping-pong
    bf16_t* __restrict__ hbb,              // [2][B*H] bf16 ping-pong
    bf16_t* __restrict__ hw_b,             // [B*H] bf16 out
    float* __restrict__ hsum_f,            // [B*H] f32 out
    float* __restrict__ spread_acc,        // [B*C] f32 out
    unsigned* __restrict__ bar) {
  const int wg = blockIdx.x;
  const int lane = threadIdx.x & 31;
  const int w = threadIdx.x >> 5;
  const int g = lane >> 4, ln = lane & 15;
  const int mt = w & 3;    // batch tile (16 rows)
  const int nw = w >> 2;   // 32-col j window within slice
  const int j0 = wg * 64;
  const int ncol0 = j0 + nw * 32;          // first hp column (per gate) for this wave

  float accW[2][8], accS[2][8];
#pragma unroll
  for (int jj = 0; jj < 2; ++jj)
#pragma unroll
    for (int v = 0; v < 8; ++v) { accW[jj][v] = 0.f; accS[jj][v] = 0.f; }
  v8f sAcc = {};  // WG0 spreads tile accumulator (wave = one 16x16 tile)

  for (int t = 0; t < T_DIM; ++t) {
    const int pr = t & 1;
    const int pw = pr ^ 1;
    const bf16_t* hRead = hbb + (size_t)pr * (B_DIM * H_DIM);

    // WG0: spread projection of h_{t-1} (published last step), decay d^(1024-t)
    if (wg == 0 && t > 0) {
      const float dk = __powf(DECAY_F, (float)(T_DIM - t));
      v8f c = {};
      for (int ks = 0; ks < 16; ++ks) {
        v16bf a = load_a_tile(hRead + (size_t)(mt * 16) * H_DIM + ks * 32, H_DIM, lane);
        v16bf b = load_b_tile(wsm + (size_t)(nw * 16) * H_DIM + ks * 32, H_DIM, lane);
        c = wmma_bf16(a, b, c);
      }
      const float bsl = bsv[nw * 16 + ln];
#pragma unroll
      for (int v = 0; v < 8; ++v) sAcc[v] += dk * sigmoidf_(c[v] + bsl);
    }

    // hp slice = h_{t-1} @ whh^T for columns {gt*512 + ncol0 .. +32}.
    // Two-stage ping-pong pipeline, fully unrolled (8 pairs), no copies.
    v8f hp[3][2] = {};
    const bf16_t* aBase = hRead + (size_t)(mt * 16) * H_DIM;
    v16bf a0 = load_a_tile(aBase, H_DIM, lane);
    v16bf a1 = load_a_tile(aBase + 32, H_DIM, lane);
    v16bf b0[3][2], b1[3][2];
#pragma unroll
    for (int gt = 0; gt < 3; ++gt)
#pragma unroll
      for (int jj = 0; jj < 2; ++jj) {
        const bf16_t* br = whh + (size_t)(gt * H_DIM + ncol0 + jj * 16) * H_DIM;
        b0[gt][jj] = load_b_tile(br, H_DIM, lane);
        b1[gt][jj] = load_b_tile(br + 32, H_DIM, lane);
      }
#pragma unroll
    for (int ks = 0; ks < 16; ks += 2) {
#pragma unroll
      for (int gt = 0; gt < 3; ++gt)
#pragma unroll
        for (int jj = 0; jj < 2; ++jj)
          hp[gt][jj] = wmma_bf16(a0, b0[gt][jj], hp[gt][jj]);
      if (ks + 2 < 16) {
        a0 = load_a_tile(aBase + (ks + 2) * 32, H_DIM, lane);
#pragma unroll
        for (int gt = 0; gt < 3; ++gt)
#pragma unroll
          for (int jj = 0; jj < 2; ++jj)
            b0[gt][jj] = load_b_tile(
                whh + (size_t)(gt * H_DIM + ncol0 + jj * 16) * H_DIM + (ks + 2) * 32,
                H_DIM, lane);
      }
#pragma unroll
      for (int gt = 0; gt < 3; ++gt)
#pragma unroll
        for (int jj = 0; jj < 2; ++jj)
          hp[gt][jj] = wmma_bf16(a1, b1[gt][jj], hp[gt][jj]);
      if (ks + 3 < 16) {
        a1 = load_a_tile(aBase + (ks + 3) * 32, H_DIM, lane);
#pragma unroll
        for (int gt = 0; gt < 3; ++gt)
#pragma unroll
          for (int jj = 0; jj < 2; ++jj)
            b1[gt][jj] = load_b_tile(
                whh + (size_t)(gt * H_DIM + ncol0 + jj * 16) * H_DIM + (ks + 3) * 32,
                H_DIM, lane);
      }
    }

    // Gates + state update + fused accumulations
    const float dk_t = __powf(DECAY_F, (float)(T_DIM - 1 - t));
#pragma unroll
    for (int jj = 0; jj < 2; ++jj) {
      const int jgl = ncol0 + jj * 16 + ln;
      const float br = bhh[jgl];
      const float bz = bhh[H_DIM + jgl];
      const float bn = bhh[2 * H_DIM + jgl];
#pragma unroll
      for (int v = 0; v < 8; ++v) {
        const int bidx = mt * 16 + v + 8 * g;
        const float* xr = xp + ((size_t)bidx * T_DIM + t) * G3H;
        // Prefetch next step's gate operands (streamed-once from HBM).
        if (t + 1 < T_DIM) {
          const float* xr1 = xr + G3H;
          __builtin_prefetch(xr1 + jgl, 0, 3);
          __builtin_prefetch(xr1 + H_DIM + jgl, 0, 3);
          __builtin_prefetch(xr1 + 2 * H_DIM + jgl, 0, 3);
        }
        const float r = sigmoidf_(xr[jgl] + hp[0][jj][v] + br);
        const float z = sigmoidf_(xr[H_DIM + jgl] + hp[1][jj][v] + bz);
        const float n = tanhf(xr[2 * H_DIM + jgl] + r * (hp[2][jj][v] + bn));
        const float hprev = hf[(size_t)pr * (B_DIM * H_DIM) + (size_t)bidx * H_DIM + jgl];
        const float hnew = (1.f - z) * n + z * hprev;
        hf[(size_t)pw * (B_DIM * H_DIM) + (size_t)bidx * H_DIM + jgl] = hnew;
        hbb[(size_t)pw * (B_DIM * H_DIM) + (size_t)bidx * H_DIM + jgl] = to_bf16(hnew);
        accW[jj][v] += dk_t * hnew;
        accS[jj][v] += hnew;
      }
    }
    grid_barrier(bar, (unsigned)t);
  }

  // WG0: spread contribution of h_{T-1} (decay = 1); T even -> buffer 0
  if (wg == 0) {
    const bf16_t* hLast = hbb;
    v8f c = {};
    for (int ks = 0; ks < 16; ++ks) {
      v16bf a = load_a_tile(hLast + (size_t)(mt * 16) * H_DIM + ks * 32, H_DIM, lane);
      v16bf b = load_b_tile(wsm + (size_t)(nw * 16) * H_DIM + ks * 32, H_DIM, lane);
      c = wmma_bf16(a, b, c);
    }
    const float bsl = bsv[nw * 16 + ln];
#pragma unroll
    for (int v = 0; v < 8; ++v) {
      sAcc[v] += sigmoidf_(c[v] + bsl);
      spread_acc[(size_t)(mt * 16 + v + 8 * g) * C_DIM + nw * 16 + ln] = sAcc[v];
    }
  }
  // hw / hsum epilogue (disjoint ownership -> plain stores)
#pragma unroll
  for (int jj = 0; jj < 2; ++jj) {
    const int jgl = ncol0 + jj * 16 + ln;
#pragma unroll
    for (int v = 0; v < 8; ++v) {
      const int bidx = mt * 16 + v + 8 * g;
      hw_b[(size_t)bidx * H_DIM + jgl] = to_bf16(accW[jj][v]);
      hsum_f[(size_t)bidx * H_DIM + jgl] = accS[jj][v];
    }
  }
}

// weights = hsum@ww^T + T*bw -> softmax; spreads = d^T + (1-d)*spread_acc;
// write both into the interleaved summary columns.
__global__ __launch_bounds__(256) void finalize_kernel(
    const float* __restrict__ hsum, const float* __restrict__ ww,
    const float* __restrict__ bw, const float* __restrict__ spread_acc,
    bf16_t* __restrict__ flat, float spreadBase) {
  const int b = blockIdx.x * 8 + (threadIdx.x >> 5);
  const int c = threadIdx.x & 31;
  const float* hs = hsum + (size_t)b * H_DIM;
  const float* wr = ww + (size_t)c * H_DIM;
  float acc = 0.f;
  for (int k = 0; k < H_DIM; ++k) acc += hs[k] * wr[k];
  float wt = acc + (float)T_DIM * bw[c];
  float mx = wt;
  for (int off = 16; off; off >>= 1) mx = fmaxf(mx, __shfl_xor(mx, off, 32));
  const float e = __expf(wt - mx);
  float s = e;
  for (int off = 16; off; off >>= 1) s += __shfl_xor(s, off, 32);
  const float nw = e / s;
  const float spread = spreadBase + (1.f - DECAY_F) * spread_acc[(size_t)b * C_DIM + c];
  flat[(size_t)b * KCLS + c * CW + H_DIM] = to_bf16(spread);
  flat[(size_t)b * KCLS + c * CW + H_DIM + 1] = to_bf16(nw);
}

// ---------------- host orchestration ----------------
extern "C" void kernel_launch(void* const* d_in, const int* in_sizes, int n_in,
                              void* d_out, int out_size, void* d_ws, size_t ws_size,
                              hipStream_t stream) {
  (void)in_sizes; (void)n_in; (void)out_size; (void)ws_size;
  const int*   tokens = (const int*)d_in[0];
  const float* emb    = (const float*)d_in[1];
  const float* w_ih   = (const float*)d_in[2];
  const float* w_hh   = (const float*)d_in[3];
  const float* b_ih   = (const float*)d_in[4];
  const float* b_hh   = (const float*)d_in[5];
  const float* wc     = (const float*)d_in[6];
  const float* bc     = (const float*)d_in[7];
  const float* wsm    = (const float*)d_in[8];
  const float* bs     = (const float*)d_in[9];
  const float* ww     = (const float*)d_in[10];
  const float* bw     = (const float*)d_in[11];
  const float* wcls   = (const float*)d_in[12];
  const float* bcls   = (const float*)d_in[13];

  char* base = (char*)d_ws;
  size_t off = 0;
  auto carve = [&](size_t bytes) -> void* {
    void* r = base + off;
    off = (off + bytes + 255) & ~(size_t)255;
    return r;
  };
  const size_t NBT  = (size_t)B_DIM * T_DIM;            // 65536
  bf16_t* xb      = (bf16_t*)carve(NBT * H_DIM * 2);          // 64 MB
  bf16_t* wih_b   = (bf16_t*)carve((size_t)G3H * H_DIM * 2);
  bf16_t* whh_b   = (bf16_t*)carve((size_t)G3H * H_DIM * 2);
  bf16_t* wc_b    = (bf16_t*)carve((size_t)C_DIM * H_DIM * H_DIM * 2);  // [16384,512]
  bf16_t* ws_b    = (bf16_t*)carve((size_t)C_DIM * H_DIM * 2);
  bf16_t* wcls_b  = (bf16_t*)carve((size_t)NCLS_PAD * KCLS * 2);
  float*  xp      = (float*)carve(NBT * G3H * 4);             // 402 MB
  float*  hf      = (float*)carve((size_t)2 * B_DIM * H_DIM * 4);
  bf16_t* hbb     = (bf16_t*)carve((size_t)2 * B_DIM * H_DIM * 2);
  bf16_t* hw_b    = (bf16_t*)carve((size_t)B_DIM * H_DIM * 2);
  float*  hsum    = (float*)carve((size_t)B_DIM * H_DIM * 4);
  float*  spread  = (float*)carve((size_t)B_DIM * C_DIM * 4);
  bf16_t* flat    = (bf16_t*)carve((size_t)B_DIM * KCLS * 2);
  unsigned* bar   = (unsigned*)carve(256);

  // 1) conversions + gather
  {
    const long long n4 = (long long)NBT * (H_DIM / 4);
    gather_cvt_kernel<<<(unsigned)((n4 + 255) / 256), 256, 0, stream>>>(tokens, emb, xb);
  }
  {
    const long long n4 = (long long)G3H * H_DIM / 4;
    cvt_bf16_kernel<<<(unsigned)((n4 + 255) / 256), 256, 0, stream>>>(w_ih, wih_b, n4);
    cvt_bf16_kernel<<<(unsigned)((n4 + 255) / 256), 256, 0, stream>>>(w_hh, whh_b, n4);
  }
  {
    const long long n4 = (long long)C_DIM * H_DIM * H_DIM / 4;
    cvt_bf16_kernel<<<(unsigned)((n4 + 255) / 256), 256, 0, stream>>>(wc, wc_b, n4);
  }
  {
    const long long n4 = (long long)C_DIM * H_DIM / 4;
    cvt_bf16_kernel<<<(unsigned)((n4 + 255) / 256), 256, 0, stream>>>(wsm, ws_b, n4);
  }
  {
    const long long n4 = (long long)NCLS_PAD * (KCLS / 4);
    cvt_pad_kernel<<<(unsigned)((n4 + 255) / 256), 256, 0, stream>>>(wcls, wcls_b,
                                                                     NCLS, KCLS, NCLS_PAD);
  }
  // 2) zero recurrence state + barrier (must reset every call)
  hipMemsetAsync(hf, 0, (size_t)2 * B_DIM * H_DIM * 4, stream);
  hipMemsetAsync(hbb, 0, (size_t)2 * B_DIM * H_DIM * 2, stream);
  hipMemsetAsync(bar, 0, 256, stream);

  // 3) xp = x @ w_ih^T + b_ih   (M=65536, N=1536, K=512) f32 out
  gemm_bf16_kernel<<<dim3(NBT / 64, G3H / 128), 256, 0, stream>>>(
      xb, wih_b, b_ih, xp, (int)NBT, G3H, H_DIM, 0, 1.f, 1.f, G3H);

  // 4) persistent GRU scan (8 WGs, device barrier per step)
  gru_scan_kernel<<<NWG_SCAN, 256, 0, stream>>>(xp, whh_b, b_hh, ws_b, bs, hf, hbb,
                                                hw_b, hsum, spread, bar);

  // 5) centers = (1-d)*(hw @ wc^T + Sd*bc) -> interleaved bf16 summary
  const float biasScaleC = (float)(1.0 - std::pow((double)DECAY_F, (double)T_DIM));
  gemm_bf16_kernel<<<dim3(1, (C_DIM * H_DIM) / 128), 256, 0, stream>>>(
      hw_b, wc_b, bc, flat, B_DIM, C_DIM * H_DIM, H_DIM, 1,
      1.f - DECAY_F, biasScaleC, C_DIM * H_DIM);

  // 6) weights softmax + spreads -> remaining summary columns
  const float spreadBase = (float)std::pow((double)DECAY_F, (double)T_DIM);
  finalize_kernel<<<B_DIM / 8, 256, 0, stream>>>(hsum, ww, bw, spread, flat, spreadBase);

  // 7) out = flat @ wcls^T + bcls  (M=64, N=1000 padded to 1024, K=16448)
  gemm_bf16_kernel<<<dim3(1, NCLS_PAD / 128), 256, 0, stream>>>(
      flat, wcls_b, bcls, d_out, B_DIM, NCLS_PAD, KCLS, 2, 1.f, 1.f, NCLS);
}
